// BEiTModel4Stage_91336774517017
// MI455X (gfx1250) — matrix operand order repaired and checked by reading, not compiled, not run on hardware
//
#include <hip/hip_runtime.h>
#include <hip/hip_bf16.h>
#include <math.h>

typedef __attribute__((ext_vector_type(16))) _Float16 v16h;
typedef __attribute__((ext_vector_type(8)))  float    v8f;

#define NLAYER 12
#define NHEAD  12
#define FDIM   768
#define FHD    64
#define MLPDIM 3072
#define NTOK   1025
#define NPAD   1056          /* 33*32: token count padded to K-step multiple */
#define RTAB   3972          /* (2*32-1)^2 + 3 */
#define EPSLN  1e-6f
#define QKSCALE 0.125f       /* 64^-0.5 */

#define CDIV(a,b) (((a)+(b)-1)/(b))

// ---------------------------------------------------------------------------
// Block-wide reduction helpers (wave32, 8 waves / 256 threads)
// ---------------------------------------------------------------------------
__device__ __forceinline__ float block_reduce(float v, float* sh, bool do_max) {
#pragma unroll
  for (int off = 16; off > 0; off >>= 1) {
    float o = __shfl_down(v, off, 32);
    v = do_max ? fmaxf(v, o) : (v + o);
  }
  const int w = threadIdx.x >> 5, lane = threadIdx.x & 31;
  __syncthreads();
  if (lane == 0) sh[w] = v;
  __syncthreads();
  v = sh[0];
#pragma unroll
  for (int i = 1; i < 8; ++i) v = do_max ? fmaxf(v, sh[i]) : (v + sh[i]);
  return v;
}

// ---------------------------------------------------------------------------
// Generic wave32 WMMA GEMM:  C[z] = A16[z] (MxK, row major) * B[z]^T (NoutxK)
//   B32:  B is fp32 (weights): register-pipelined load + cvt + ds_store.
//         A is staged with async global->LDS, double buffered (ASYNCcnt).
//   !B32: both A and B staged via async global->LDS, double buffered.
//   EPI:  0 = store f32, 1 = store f16, 2 = (+bias, exact GELU) -> f16
// Workgroup: 256 threads = 8 waves, 128x64 C tile, 32-wide K panels.
// Wave grid 4x2; each wave owns a 32x32 sub-tile = 4 accumulators,
// 4 v_wmma per K panel with 2x fragment reuse.
// Async issue counts are uniform per wave (OOB lanes clamped to a dummy LDS
// chunk + safe global row) so in-order ASYNCcnt waits pipeline correctly.
// ---------------------------------------------------------------------------
#define BLKM 128
#define BLKN 64
#define BKK  32
#define LDST 40   /* LDS row stride in halves: 80B, 16B-aligned chunks */

template<bool B32, int EPI>
__global__ __launch_bounds__(256) void wmma_gemm(
    const _Float16* __restrict__ A, const void* __restrict__ Bv,
    const float* __restrict__ bias, void* __restrict__ Cv,
    int M, int Nout, int K, int lda, int ldb, int ldc,
    long long sA, long long sB, long long sC)
{
  __shared__ __align__(16) _Float16 As[2][BLKM * LDST];
  __shared__ __align__(16) _Float16 Bs[2][BLKN * LDST];
  __shared__ __align__(16) _Float16 dummy[8];   // clamp target for OOB lanes

  const int tid  = threadIdx.x;
  const int lane = tid & 31;
  const int w    = tid >> 5;       // wave 0..7
  const int wm   = w >> 1;         // 0..3  -> 32-row strip
  const int wn   = w & 1;          // 0..1  -> 32-col strip
  const int bm   = blockIdx.y * BLKM;
  const int bn   = blockIdx.x * BLKN;
  const int z    = blockIdx.z;

  const _Float16* Ab = A + (long long)z * sA;
  const float*    Bf = (const float*)Bv    + (long long)z * sB;
  const _Float16* Bh = (const _Float16*)Bv + (long long)z * sB;

  v8f acc00 = {0.f,0.f,0.f,0.f,0.f,0.f,0.f,0.f};
  v8f acc01 = acc00, acc10 = acc00, acc11 = acc00;

  // 16B-chunk ownership: A panel 128x32 halves = 512 chunks (2/thread),
  // B panel 64x32 halves = 256 chunks (1/thread).
  const int cr = tid >> 2;          // chunk row 0..63
  const int ck = (tid & 3) * 8;     // k offset in halves: 0,8,16,24

  const bool okA0 = (bm + cr)      < M;
  const bool okA1 = (bm + cr + 64) < M;
  const bool okB  = (bn + cr)      < Nout;

  // Pre-zero OOB-owned chunks in both buffers (async never writes them).
  {
    const uint4 z4 = {0u, 0u, 0u, 0u};
    if (!okA0) { *(uint4*)&As[0][cr*LDST+ck] = z4; *(uint4*)&As[1][cr*LDST+ck] = z4; }
    if (!okA1) { *(uint4*)&As[0][(cr+64)*LDST+ck] = z4; *(uint4*)&As[1][(cr+64)*LDST+ck] = z4; }
    if (!B32 && !okB) { *(uint4*)&Bs[0][cr*LDST+ck] = z4; *(uint4*)&Bs[1][cr*LDST+ck] = z4; }
  }

  // Per-lane clamped global source pointers; advance 32 halves per panel.
  const _Float16* gA0 = Ab + (okA0 ? (long long)(bm + cr) * lda : 0) + ck;
  const _Float16* gA1 = Ab + (okA1 ? (long long)(bm + cr + 64) * lda : 0) + ck;
  const _Float16* gB  = Bh + ((!B32 && okB) ? (long long)(bn + cr) * ldb : 0) + ck;
  const float*  pB32  = Bf + (okB ? (long long)(bn + cr) * ldb : 0) + ck;

  const unsigned dDum = (unsigned)(unsigned long long)(const void*)&dummy[0];
  unsigned dA0[2], dA1[2], dB[2];
#pragma unroll
  for (int b = 0; b < 2; ++b) {
    dA0[b] = okA0 ? (unsigned)(unsigned long long)(const void*)&As[b][cr*LDST+ck] : dDum;
    dA1[b] = okA1 ? (unsigned)(unsigned long long)(const void*)&As[b][(cr+64)*LDST+ck] : dDum;
    dB[b]  = okB  ? (unsigned)(unsigned long long)(const void*)&Bs[b][cr*LDST+ck] : dDum;
  }

  // Uniform-count async issue of one K panel into buffer `b` (2 or 3 ops).
  auto issue_panel = [&](int b) {
    asm volatile("global_load_async_to_lds_b128 %0, %1, off"
                 :: "v"(dA0[b]), "v"((unsigned long long)(const void*)gA0) : "memory");
    asm volatile("global_load_async_to_lds_b128 %0, %1, off"
                 :: "v"(dA1[b]), "v"((unsigned long long)(const void*)gA1) : "memory");
    if (!B32) {
      asm volatile("global_load_async_to_lds_b128 %0, %1, off"
                   :: "v"(dB[b]), "v"((unsigned long long)(const void*)gB) : "memory");
      gB += BKK;
    }
    gA0 += BKK;
    gA1 += BKK;
  };

  // fp32-weight register pipeline (B32 path).
  float breg[8];
  auto loadB32 = [&]() {
    if (okB) {
#pragma unroll
      for (int i = 0; i < 8; ++i) breg[i] = pB32[i];
      __builtin_prefetch(pB32 + BKK, 0, 1);
    } else {
#pragma unroll
      for (int i = 0; i < 8; ++i) breg[i] = 0.f;
    }
    pB32 += BKK;
  };
  auto storeB32 = [&](int b) {
    _Float16 t[8];
#pragma unroll
    for (int i = 0; i < 8; ++i) t[i] = (_Float16)breg[i];
#pragma unroll
    for (int i = 0; i < 8; ++i) Bs[b][cr*LDST+ck+i] = t[i];
  };

  const int nk = K / BKK;   // K always a multiple of 32; nk >= 2 here

  // Prologue: two panels in flight.
  issue_panel(0);
  if (nk > 1) issue_panel(1);
  if (B32) loadB32();       // panel 0 weights into registers

  const int half = lane >> 4;      // 0/1
  const int l16  = lane & 15;

  for (int i = 0; i < nk; ++i) {
    const int buf = i & 1;
    if (B32) storeB32(buf);  // panel i weights -> LDS (buf last read in iter i-2)

    // Drain to the depth of the newer in-flight panel (in-order completion).
    if (i + 1 < nk) {
      if (B32) asm volatile("s_wait_asynccnt 0x2" ::: "memory");
      else     asm volatile("s_wait_asynccnt 0x3" ::: "memory");
    } else {
      asm volatile("s_wait_asynccnt 0x0" ::: "memory");
    }
    __syncthreads();

    if (B32 && i + 1 < nk) loadB32();   // overlap next weight fetch with math

    // ---- fragments (contiguous k = 16*half + e, same order in A and B) ----
    v16h a0, a1, b0, b1;
    {
      const _Float16* ar0 = &As[buf][(wm * 32 + l16) * LDST];
      const _Float16* ar1 = ar0 + 16 * LDST;
#pragma unroll
      for (int e = 0; e < 16; ++e) {
        const int kk = e + 16 * half;
        a0[e] = ar0[kk];
        a1[e] = ar1[kk];
      }
      const _Float16* br0 = &Bs[buf][(wn * 32 + l16) * LDST + 16 * half];
      const _Float16* br1 = br0 + 16 * LDST;
#pragma unroll
      for (int e = 0; e < 16; ++e) { b0[e] = br0[e]; b1[e] = br1[e]; }
    }

    acc00 = __builtin_amdgcn_wmma_f32_16x16x32_f16(false, a0, false, b0,
                                                   (short)0, acc00, false, false);
    acc01 = __builtin_amdgcn_wmma_f32_16x16x32_f16(false, a0, false, b1,
                                                   (short)0, acc01, false, false);
    acc10 = __builtin_amdgcn_wmma_f32_16x16x32_f16(false, a1, false, b0,
                                                   (short)0, acc10, false, false);
    acc11 = __builtin_amdgcn_wmma_f32_16x16x32_f16(false, a1, false, b1,
                                                   (short)0, acc11, false, false);
    __syncthreads();         // all reads of `buf` done before it is refilled

    if (i + 2 < nk) issue_panel(buf);   // panel i+2 -> same buffer
  }

  // ---- epilogue: D layout -> lane col n=l16, VGPR r row m = r + 8*half ----
  const int n0 = bn + wn * 32 + l16;
  const int n1 = n0 + 16;
  const int mb0 = bm + wm * 32 + 8 * half;
  const int mb1 = mb0 + 16;

  auto store_acc = [&](const v8f& acc, int mbase, int n) {
    if (n >= Nout) return;
#pragma unroll
    for (int r = 0; r < 8; ++r) {
      const int m = mbase + r;
      if (m >= M) continue;
      if (EPI == 0) {
        float* C = (float*)Cv + (long long)z * sC;
        C[(long long)m * ldc + n] = acc[r];
      } else if (EPI == 1) {
        _Float16* C = (_Float16*)Cv + (long long)z * sC;
        C[(long long)m * ldc + n] = (_Float16)acc[r];
      } else {
        _Float16* C = (_Float16*)Cv + (long long)z * sC;
        const float v = acc[r] + bias[n];
        C[(long long)m * ldc + n] =
            (_Float16)(0.5f * v * (1.f + erff(v * 0.70710678118f)));
      }
    }
  };
  store_acc(acc00, mb0, n0);
  store_acc(acc01, mb0, n1);
  store_acc(acc10, mb1, n0);
  store_acc(acc11, mb1, n1);
}

// ---------------------------------------------------------------------------
// Elementwise / reduction kernels
// ---------------------------------------------------------------------------
__global__ void concat_kernel(const float* __restrict__ patches,
                              const float* __restrict__ cls,
                              float* __restrict__ x)
{
  const int i = blockIdx.x * 256 + threadIdx.x;
  if (i >= NTOK * FDIM) return;
  const int n = i / FDIM, f = i - n * FDIM;
  x[i] = (n == 0) ? cls[f] : patches[(n - 1) * FDIM + f];
}

__global__ __launch_bounds__(256) void ln_kernel(
    const float* __restrict__ x, const float* __restrict__ w,
    const float* __restrict__ b, _Float16* __restrict__ out)
{
  const int n = blockIdx.x;
  const float* row = x + (long long)n * FDIM;
  float v[3], s = 0.f, s2 = 0.f;
#pragma unroll
  for (int i = 0; i < 3; ++i) {
    const int f = threadIdx.x + i * 256;
    v[i] = row[f];
    s += v[i];
    s2 += v[i] * v[i];
  }
  __shared__ float sh[8];
  s  = block_reduce(s,  sh, false);
  s2 = block_reduce(s2, sh, false);
  const float mu = s * (1.f / FDIM);
  const float var = s2 * (1.f / FDIM) - mu * mu;
  const float rs = rsqrtf(var + EPSLN);
#pragma unroll
  for (int i = 0; i < 3; ++i) {
    const int f = threadIdx.x + i * 256;
    out[(long long)n * FDIM + f] = (_Float16)((v[i] - mu) * rs * w[f] + b[f]);
  }
}

__global__ void qkv_split_kernel(const float* __restrict__ qkv,
                                 const float* __restrict__ qb,
                                 const float* __restrict__ vb,
                                 _Float16* __restrict__ q16,
                                 _Float16* __restrict__ k16,
                                 _Float16* __restrict__ vT16)
{
  const long long i = (long long)blockIdx.x * 256 + threadIdx.x;
  if (i >= (long long)NHEAD * NPAD * FHD) return;
  const int d = (int)(i % FHD);
  const int n = (int)((i / FHD) % NPAD);
  const int h = (int)(i / ((long long)FHD * NPAD));
  float qv = 0.f, kv = 0.f, vv = 0.f;
  if (n < NTOK) {
    const float* r = qkv + (long long)n * (3 * FDIM);
    qv = (r[h * FHD + d] + qb[h * FHD + d]) * QKSCALE;
    kv = r[FDIM + h * FHD + d];
    vv = r[2 * FDIM + h * FHD + d] + vb[h * FHD + d];
  }
  q16[((long long)h * NPAD + n) * FHD + d]  = (_Float16)qv;
  k16[((long long)h * NPAD + n) * FHD + d]  = (_Float16)kv;
  vT16[((long long)h * FHD + d) * NPAD + n] = (_Float16)vv;
}

__global__ __launch_bounds__(256) void softmax_kernel(
    const float* __restrict__ S, const float* __restrict__ table,
    const int* __restrict__ ridx, _Float16* __restrict__ P)
{
  const int n = blockIdx.x;   // 0..NTOK-1
  const int h = blockIdx.y;   // 0..NHEAD-1
  const float* srow = S + ((long long)h * NPAD + n) * NPAD;
  _Float16*    prow = P + ((long long)h * NPAD + n) * NPAD;
  const int*   irow = ridx + (long long)n * NTOK;

  float sv[5];
  float mx = -1e30f;
#pragma unroll
  for (int i = 0; i < 5; ++i) {
    const int m = threadIdx.x + i * 256;
    float v = -1e30f;
    if (m < NTOK) v = srow[m] + table[(long long)irow[m] * NHEAD + h];
    sv[i] = v;
    mx = fmaxf(mx, v);
  }
  __shared__ float sh[8];
  mx = block_reduce(mx, sh, true);
  float sum = 0.f;
#pragma unroll
  for (int i = 0; i < 5; ++i) {
    const int m = threadIdx.x + i * 256;
    if (m < NTOK) { sv[i] = __expf(sv[i] - mx); sum += sv[i]; }
  }
  sum = block_reduce(sum, sh, false);
  const float inv = 1.f / sum;
#pragma unroll
  for (int i = 0; i < 5; ++i) {
    const int m = threadIdx.x + i * 256;
    if (m < NPAD) prow[m] = (_Float16)((m < NTOK) ? sv[i] * inv : 0.f);
  }
}

__global__ void residual_kernel(float* __restrict__ x,
                                const float* __restrict__ y,
                                const float* __restrict__ bias,
                                const float* __restrict__ scale)
{
  const int i = blockIdx.x * 256 + threadIdx.x;
  if (i >= NTOK * FDIM) return;
  const int f = i % FDIM;
  x[i] += scale[f] * (y[i] + bias[f]);
}

// ---------------------------------------------------------------------------
// Host orchestration
// ---------------------------------------------------------------------------
extern "C" void kernel_launch(void* const* d_in, const int* in_sizes, int n_in,
                              void* d_out, int out_size, void* d_ws, size_t ws_size,
                              hipStream_t stream) {
  const float* patch_tokens = (const float*)d_in[0];
  const float* cls_token    = (const float*)d_in[1];
  const float* ln1_w        = (const float*)d_in[2];
  const float* ln1_b        = (const float*)d_in[3];
  const float* qkv_w        = (const float*)d_in[4];
  const float* q_bias       = (const float*)d_in[5];
  const float* v_bias       = (const float*)d_in[6];
  const float* proj_w       = (const float*)d_in[7];
  const float* proj_b       = (const float*)d_in[8];
  const float* scale_attn   = (const float*)d_in[9];
  const float* ln2_w        = (const float*)d_in[10];
  const float* ln2_b        = (const float*)d_in[11];
  const float* fc1_w        = (const float*)d_in[12];
  const float* fc1_b        = (const float*)d_in[13];
  const float* fc2_w        = (const float*)d_in[14];
  const float* fc2_b        = (const float*)d_in[15];
  const float* scale_mlp    = (const float*)d_in[16];
  const float* relpos_table = (const float*)d_in[17];
  const int*   relpos_index = (const int*)d_in[18];
  float* out = (float*)d_out;

  // ---- workspace arena ----
  char* ws = (char*)d_ws;
  size_t off = 0;
  auto alloc = [&](size_t bytes) {
    void* p = ws + off;
    off += (bytes + 255) & ~(size_t)255;
    return p;
  };
  float*    x     = (float*)   alloc(sizeof(float)    * NTOK * FDIM);
  _Float16* h16   = (_Float16*)alloc(sizeof(_Float16) * NTOK * FDIM);
  float*    qkv   = (float*)   alloc(sizeof(float)    * NTOK * 3 * FDIM);
  _Float16* q16   = (_Float16*)alloc(sizeof(_Float16) * NHEAD * NPAD * FHD);
  _Float16* k16   = (_Float16*)alloc(sizeof(_Float16) * NHEAD * NPAD * FHD);
  _Float16* vT16  = (_Float16*)alloc(sizeof(_Float16) * NHEAD * FHD * NPAD);
  float*    S     = (float*)   alloc(sizeof(float)    * (size_t)NHEAD * NPAD * NPAD);
  _Float16* P16   = (_Float16*)alloc(sizeof(_Float16) * (size_t)NHEAD * NPAD * NPAD);
  _Float16* O16   = (_Float16*)alloc(sizeof(_Float16) * NTOK * FDIM);
  float*    tmp   = (float*)   alloc(sizeof(float)    * NTOK * FDIM);
  _Float16* m116  = (_Float16*)alloc(sizeof(_Float16) * NTOK * MLPDIM);
  (void)ws_size; (void)n_in; (void)in_sizes; (void)out_size;

  const int ew_blocks = CDIV(NTOK * FDIM, 256);
  const int mt = CDIV(NTOK, BLKM);   // 9 M-tiles of 128

  concat_kernel<<<ew_blocks, 256, 0, stream>>>(patch_tokens, cls_token, x);

  int stage = 0;
  for (int l = 0; l < NLAYER; ++l) {
    const float* qkv_w_l = qkv_w + (long long)l * 3 * FDIM * FDIM;
    const float* prj_w_l = proj_w + (long long)l * FDIM * FDIM;
    const float* fc1_w_l = fc1_w + (long long)l * MLPDIM * FDIM;
    const float* fc2_w_l = fc2_w + (long long)l * FDIM * MLPDIM;

    // LN1 -> h16
    ln_kernel<<<NTOK, 256, 0, stream>>>(x, ln1_w + l * FDIM, ln1_b + l * FDIM, h16);

    // qkv = h16 @ qkv_w^T      (M=1025, N=2304, K=768)
    wmma_gemm<true, 0><<<dim3(CDIV(3 * FDIM, BLKN), mt, 1), 256, 0, stream>>>(
        h16, qkv_w_l, nullptr, qkv, NTOK, 3 * FDIM, FDIM, FDIM, FDIM, 3 * FDIM, 0, 0, 0);

    // split + biases + scale + transpose V, with zero K-padding
    qkv_split_kernel<<<CDIV(NHEAD * NPAD * FHD, 256), 256, 0, stream>>>(
        qkv, q_bias + l * NHEAD * FHD, v_bias + l * NHEAD * FHD, q16, k16, vT16);

    // S[h] = q16[h] @ k16[h]^T  (batched over heads, K=64)
    wmma_gemm<false, 0><<<dim3(CDIV(NTOK, BLKN), mt, NHEAD), 256, 0, stream>>>(
        q16, k16, nullptr, S, NTOK, NTOK, FHD, FHD, FHD, NPAD,
        (long long)NPAD * FHD, (long long)NPAD * FHD, (long long)NPAD * NPAD);

    // softmax(S + relpos_bias) -> P16 (f16, zero-padded K dim)
    softmax_kernel<<<dim3(NTOK, NHEAD), 256, 0, stream>>>(
        S, relpos_table + (long long)l * RTAB * NHEAD, relpos_index, P16);

    // O[:, h*64:(h+1)*64] = P16[h] @ vT16[h]^T  (K=1056) -> f16 [N,768]
    wmma_gemm<false, 1><<<dim3(1, mt, NHEAD), 256, 0, stream>>>(
        P16, vT16, nullptr, O16, NTOK, FHD, NPAD, NPAD, NPAD, FDIM,
        (long long)NPAD * NPAD, (long long)FHD * NPAD, (long long)FHD);

    // proj: tmp = O16 @ proj_w^T
    wmma_gemm<true, 0><<<dim3(CDIV(FDIM, BLKN), mt, 1), 256, 0, stream>>>(
        O16, prj_w_l, nullptr, tmp, NTOK, FDIM, FDIM, FDIM, FDIM, FDIM, 0, 0, 0);

    residual_kernel<<<ew_blocks, 256, 0, stream>>>(
        x, tmp, proj_b + l * FDIM, scale_attn + l * FDIM);

    // LN2 -> h16
    ln_kernel<<<NTOK, 256, 0, stream>>>(x, ln2_w + l * FDIM, ln2_b + l * FDIM, h16);

    // fc1 + bias + GELU -> m116 (f16)
    wmma_gemm<true, 2><<<dim3(CDIV(MLPDIM, BLKN), mt, 1), 256, 0, stream>>>(
        h16, fc1_w_l, fc1_b + (long long)l * MLPDIM, m116,
        NTOK, MLPDIM, FDIM, FDIM, FDIM, MLPDIM, 0, 0, 0);

    // fc2: tmp = m116 @ fc2_w^T  (K=3072)
    wmma_gemm<true, 0><<<dim3(CDIV(FDIM, BLKN), mt, 1), 256, 0, stream>>>(
        m116, fc2_w_l, nullptr, tmp, NTOK, FDIM, MLPDIM, MLPDIM, MLPDIM, FDIM, 0, 0, 0);

    residual_kernel<<<ew_blocks, 256, 0, stream>>>(
        x, tmp, fc2_b + l * FDIM, scale_mlp + l * FDIM);

    if (l == 2 || l == 5 || l == 8 || l == 11) {
      hipMemcpyAsync(out + (size_t)stage * NTOK * FDIM, x,
                     sizeof(float) * NTOK * FDIM, hipMemcpyDeviceToDevice, stream);
      ++stage;
    }
  }
}
